// SO2Linear_68642167324713
// MI455X (gfx1250) — matrix operand order repaired and checked by reading, not compiled, not run on hardware
//
#include <hip/hip_runtime.h>
#include <hip/hip_bf16.h>

typedef __attribute__((ext_vector_type(16))) __bf16 v16bf;
typedef __attribute__((ext_vector_type(8)))  __bf16 v8bf;
typedef __attribute__((ext_vector_type(8)))  float  v8f;

// ---------------------------------------------------------------------------
// Constants for LMAX=3, MUL=64:
//   per-l block offsets OFF[l] = {0,64,256,576}
//   C_M = {256,192,128,64}, WSL = {0,256,448,576,640}
//   gathered layout (bf16, 1024 cols): [m0:256][m1s0:192][m1s1:192]
//                                      [m2s0:128][m2s1:128][m3s0:64][m3s1:64]
//   packed-B regions (halfs): m0 @0 (65536), m1 @65536 (147456),
//                             m2 @212992 (65536), m3 @278528 (16384)
// ---------------------------------------------------------------------------

// Kernel 1: combined bf16 weight matrices, pre-swizzled into the WMMA
// 16x16x32 bf16 B operand, in this file's contiguous-K order (per 32x16 tile: lane L holds column
// N=L%16, K = (L/16)*16 + h for halfs h=0..15, stored contiguously).
__global__ void so2_prep_weights(const float* __restrict__ W0,
                                 const float* __restrict__ W1,
                                 const float* __restrict__ W2,
                                 const float* __restrict__ W3,
                                 __bf16* __restrict__ pb) {
  int tid = blockIdx.x * blockDim.x + threadIdx.x;
  if (tid >= 294912) return;
  int m, rel;
  if      (tid < 65536)  { m = 0; rel = tid; }
  else if (tid < 212992) { m = 1; rel = tid - 65536; }
  else if (tid < 278528) { m = 2; rel = tid - 212992; }
  else                   { m = 3; rel = tid - 278528; }
  const int CM[4] = {256, 192, 128, 64};
  int c    = CM[m];
  int Ntot = (m == 0) ? 256 : (2 * c);
  int tile = rel >> 9;
  int lane = (rel >> 4) & 31;
  int h    = rel & 15;
  int ntiles = Ntot >> 4;
  int kt = tile / ntiles;
  int nt = tile - kt * ntiles;
  int k = kt * 32 + (lane >> 4) * 16 + h;   // input dim
  int n = nt * 16 + (lane & 15);            // output dim
  float v;
  if (m == 0) {
    v = W0[n * 256 + k];                    // y0 = x0 @ W0^T  ->  Wc[k,n] = W0[n,k]
  } else {
    const float* Wm = (m == 1) ? W1 : ((m == 2) ? W2 : W3);
    // Wm is (2c, c) row-major; A = Wm[:c], B = Wm[c:]
    // Wc = [[A^T, -B^T], [B^T, A^T]]  so that  [y_neg|y_pos] = [x0 x1] @ Wc
    if (k < c) v = (n < c) ?  Wm[n * c + k]           : -Wm[n * c + k];
    else       v = (n < c) ?  Wm[(c + n) * c + (k-c)] :  Wm[(n - c) * c + (k-c)];
  }
  pb[tid] = (__bf16)v;
}

// Kernel 2: forward rotation r[c,i] = sum_j x[c,j] * D[j,i], gathered into
// m-grouped bf16 layout for the GEMMs. One thread per (edge, element).
__global__ void so2_rot_fwd_gather(const float* __restrict__ x,
                                   const float* __restrict__ D0,
                                   const float* __restrict__ D1,
                                   const float* __restrict__ D2,
                                   const float* __restrict__ D3,
                                   __bf16* __restrict__ xg) {
  long gid = (long)blockIdx.x * blockDim.x + threadIdx.x;
  int z   = (int)(gid >> 10);
  int col = (int)(gid & 1023);
  int l, rem;
  if      (col < 64)  { l = 0; rem = col; }
  else if (col < 256) { l = 1; rem = col - 64; }
  else if (col < 576) { l = 2; rem = col - 256; }
  else                { l = 3; rem = col - 576; }
  int n = 2 * l + 1;
  int c = rem / n;
  int i = rem - c * n;
  const int OFF[4] = {0, 64, 256, 576};
  const float* Dl = (l == 0) ? D0 : ((l == 1) ? D1 : ((l == 2) ? D2 : D3));
  const float* Dz = Dl + (size_t)z * n * n;
  const float* xr = x + ((size_t)z << 10) + OFF[l] + c * n;
  float s = 0.f;
  for (int j = 0; j < n; ++j) s += xr[j] * Dz[j * n + i];
  int mp = i - l;
  int g;
  if (mp == 0) {
    g = l * 64 + c;                                   // m=0 block, l-major
  } else {
    int mm = mp > 0 ? mp : -mp;
    const int XG[4] = {0, 256, 640, 896};
    const int CM[4] = {256, 192, 128, 64};
    g = XG[mm] + (mp > 0 ? CM[mm] : 0) + (l - mm) * 64 + c;
  }
  xg[((size_t)z << 10) + g] = (__bf16)s;
}

// Kernel 3: WMMA bf16 GEMM + SO2 epilogue. One wave computes a 16x64 output
// strip (4 N-tiles), reusing each A fragment for 4 WMMAs per k-step; then
// applies bias/edge-weight/mask and scatters f32 results into outS.
__global__ void so2_gemm_mix(const __bf16* __restrict__ xg,
                             const __bf16* __restrict__ pbR,
                             const float*  __restrict__ bias,
                             const float*  __restrict__ weight,
                             const unsigned char* __restrict__ mask,
                             float* __restrict__ outS,
                             int Ktot, int Ntot, int aOff, int wOff, int mVal) {
  int lane  = threadIdx.x;           // 0..31, wave32, EXEC all ones
  int mtile = blockIdx.x;
  int nbase = blockIdx.y * 4;        // 4 consecutive N-tiles per wave

  v8f acc0 = {0.f,0.f,0.f,0.f,0.f,0.f,0.f,0.f};
  v8f acc1 = acc0, acc2 = acc0, acc3 = acc0;

  int row = mtile * 16 + (lane & 15);
  // A layout: lane<16 -> K {0..15}; lane>=16 -> K {16..31}
  const __bf16* arow = xg + ((size_t)row << 10) + aOff + ((lane >> 4) * 16);
  int nkt    = Ktot >> 5;
  int ntiles = Ntot >> 4;
  const __bf16* bbase = pbR + ((size_t)nbase << 9) + lane * 16;

  for (int kt = 0; kt < nkt; ++kt) {
    v8bf a0 = *(const v8bf*)(arow + kt * 32);
    v8bf a1 = *(const v8bf*)(arow + kt * 32 + 8);
    __builtin_prefetch(arow + (kt + 1) * 32, 0, 0);   // global_prefetch_b8
    v16bf A = __builtin_shufflevector(a0, a1, 0,1,2,3,4,5,6,7,8,9,10,11,12,13,14,15);
    const __bf16* bk = bbase + (((size_t)kt * ntiles) << 9);
    v16bf B0 = *(const v16bf*)(bk);
    v16bf B1 = *(const v16bf*)(bk + 512);
    v16bf B2 = *(const v16bf*)(bk + 1024);
    v16bf B3 = *(const v16bf*)(bk + 1536);
    acc0 = __builtin_amdgcn_wmma_f32_16x16x32_bf16(false, A, false, B0, (short)0, acc0, false, false);
    acc1 = __builtin_amdgcn_wmma_f32_16x16x32_bf16(false, A, false, B1, (short)0, acc1, false, false);
    acc2 = __builtin_amdgcn_wmma_f32_16x16x32_bf16(false, A, false, B2, (short)0, acc2, false, false);
    acc3 = __builtin_amdgcn_wmma_f32_16x16x32_bf16(false, A, false, B3, (short)0, acc3, false, false);
  }

  const int OFF[4] = {0, 64, 256, 576};
  int cc = Ntot >> 1;
  for (int r = 0; r < 8; ++r) {
    int z = mtile * 16 + ((lane >> 4) * 8) + r;   // C/D layout: M = r + 8*(lane/16)
    float me = mask[z] ? 1.f : 0.f;
    const float* wrow = weight + (size_t)z * 640;
    float* orow = outS + ((size_t)z << 10);
    for (int sub = 0; sub < 4; ++sub) {
      int ncol = (nbase + sub) * 16 + (lane & 15);
      float v = (sub == 0) ? acc0[r] : (sub == 1) ? acc1[r] : (sub == 2) ? acc2[r] : acc3[r];
      float w;
      int sIdx;
      if (mVal == 0) {
        v += bias[ncol];
        w = wrow[ncol];
        int l = ncol >> 6, ch = ncol & 63;
        sIdx = OFF[l] + ch * (2 * l + 1) + l;
      } else {
        int wj = (ncol < cc) ? ncol : (ncol - cc);
        w = wrow[wOff + wj];
        int p = ncol >> 1;                 // sorted-IDX pair index
        int l = mVal + (p >> 6), ch = p & 63;
        sIdx = OFF[l] + ch * (2 * l + 1) + l + ((ncol & 1) ? mVal : -mVal);
      }
      orow[sIdx] = v * w * me;
    }
  }
}

// Kernel 4: in-place inverse rotation of d_out. 8 edges per 256-thread block,
// staged through LDS so the read-before-write is race-free.
__global__ void so2_rot_inv_inplace(float* __restrict__ out,
                                    const float* __restrict__ D0,
                                    const float* __restrict__ D1,
                                    const float* __restrict__ D2,
                                    const float* __restrict__ D3) {
  __shared__ float lds[8192];
  size_t base = (size_t)blockIdx.x * 8192;
  for (int k = 0; k < 32; ++k) {
    int idx = threadIdx.x + k * 256;
    lds[idx] = out[base + idx];
  }
  __syncthreads();
  const int OFF[4] = {0, 64, 256, 576};
  int zbase = blockIdx.x * 8;
  for (int k = 0; k < 32; ++k) {
    int idx = threadIdx.x + k * 256;
    int ze = idx >> 10, col = idx & 1023;
    int l, rem;
    if      (col < 64)  { l = 0; rem = col; }
    else if (col < 256) { l = 1; rem = col - 64; }
    else if (col < 576) { l = 2; rem = col - 256; }
    else                { l = 3; rem = col - 576; }
    int n = 2 * l + 1;
    int c = rem / n;
    int i = rem - c * n;
    const float* Dl = (l == 0) ? D0 : ((l == 1) ? D1 : ((l == 2) ? D2 : D3));
    const float* Dz = Dl + (size_t)(zbase + ze) * n * n + i * n;  // inverse: D[i,j]
    const float* b = &lds[(ze << 10) + OFF[l] + c * n];
    float s = 0.f;
    for (int j = 0; j < n; ++j) s += b[j] * Dz[j];
    out[base + idx] = s;
  }
}

extern "C" void kernel_launch(void* const* d_in, const int* in_sizes, int n_in,
                              void* d_out, int out_size, void* d_ws, size_t ws_size,
                              hipStream_t stream) {
  const float* x      = (const float*)d_in[0];
  const float* D0     = (const float*)d_in[1];
  const float* D1     = (const float*)d_in[2];
  const float* D2     = (const float*)d_in[3];
  const float* D3     = (const float*)d_in[4];
  const float* weight = (const float*)d_in[5];
  const unsigned char* mask = (const unsigned char*)d_in[6];   // jax bool -> 1 byte
  const float* W0     = (const float*)d_in[7];
  const float* b0     = (const float*)d_in[8];
  const float* W1     = (const float*)d_in[9];
  const float* W2     = (const float*)d_in[10];
  const float* W3     = (const float*)d_in[11];
  float* out = (float*)d_out;

  int E = in_sizes[0] / 1024;   // 65536

  // workspace: gathered bf16 x (E*1024*2 B) then packed bf16 weights (~576 KB)
  __bf16* xg = (__bf16*)d_ws;
  __bf16* pb = (__bf16*)((char*)d_ws + (size_t)E * 1024 * 2);

  so2_prep_weights<<<(294912 + 255) / 256, 256, 0, stream>>>(W0, W1, W2, W3, pb);
  so2_rot_fwd_gather<<<(E * 1024) / 256, 256, 0, stream>>>(x, D0, D1, D2, D3, xg);

  int mt = E / 16;  // 4096 row tiles; grid.y = (N/16)/4 strips
  so2_gemm_mix<<<dim3(mt, 4), 32, 0, stream>>>(xg, pb + 0,      b0,      weight, mask, out, 256, 256,   0,   0, 0);
  so2_gemm_mix<<<dim3(mt, 6), 32, 0, stream>>>(xg, pb + 65536,  nullptr, weight, mask, out, 384, 384, 256, 256, 1);
  so2_gemm_mix<<<dim3(mt, 4), 32, 0, stream>>>(xg, pb + 212992, nullptr, weight, mask, out, 256, 256, 640, 448, 2);
  so2_gemm_mix<<<dim3(mt, 2), 32, 0, stream>>>(xg, pb + 278528, nullptr, weight, mask, out, 128, 128, 896, 576, 3);

  so2_rot_inv_inplace<<<E / 8, 256, 0, stream>>>(out, D0, D1, D2, D3);
}
